// MultiHeadSelfAttention_60687887892740
// MI455X (gfx1250) — matrix operand order repaired and checked by reading, not compiled, not run on hardware
//
#include <hip/hip_runtime.h>
#include <math.h>

typedef __bf16 v16bf __attribute__((ext_vector_type(16)));
typedef float  v8f   __attribute__((ext_vector_type(8)));
typedef unsigned int u32x4 __attribute__((ext_vector_type(4)));
typedef unsigned int u32x8 __attribute__((ext_vector_type(8)));

// Problem dims (fixed by the reference)
constexpr int D_MODEL  = 1024;
constexpr int N_HEADS  = 16;
constexpr int HEAD_DIM = 64;
constexpr int BATCH    = 4;
constexpr int SEQ      = 2048;
constexpr int MROWS    = BATCH * SEQ;   // 8192

__device__ __forceinline__ unsigned short f2bf(float f) {
  unsigned int u = __float_as_uint(f);
  u += 0x7FFFu + ((u >> 16) & 1u);      // round-to-nearest-even
  return (unsigned short)(u >> 16);
}

// Async global->LDS copy of 16 bytes per lane (CDNA5 GLOBAL_LOAD_ASYNC_TO_LDS_B128,
// tracked by ASYNCcnt). ldsPtr's low 32 bits are the workgroup-relative LDS offset.
__device__ __forceinline__ void async_copy_b128(void* ldsPtr, const void* gptr) {
  unsigned lds_off = (unsigned)(uintptr_t)ldsPtr;
  asm volatile("global_load_async_to_lds_b128 %0, %1, off"
               :: "v"(lds_off), "v"(gptr) : "memory");
}

__device__ __forceinline__ void wait_async0() {
  asm volatile("s_wait_asynccnt 0x0" ::: "memory");
}

// TDM: DMA a 2D bf16 tile (64 x 32 rows, row stride 1024 elems) from global to
// LDS, hardware-padding each 128B row by 16B so LDS rows are 72 halves (bank-
// conflict-free). One instruction per chunk, tracked by TENSORcnt. D# layout
// per CDNA5 ISA ch.8: group0 = {flags, lds_addr, global_addr, type=2},
// group1 = {data_size/pad, tensor dims, tile dims, strides}.
__device__ __forceinline__ void tdm_load_k_chunk(unsigned lds_off, const void* gaddr) {
  const unsigned long long ga = (unsigned long long)(uintptr_t)gaddr;
  u32x4 g0;
  g0[0] = 1u;                                                  // count=1 (valid user D#)
  g0[1] = lds_off;                                             // lds_addr [63:32]
  g0[2] = (unsigned)ga;                                        // global_addr [95:64]
  g0[3] = (unsigned)((ga >> 32) & 0x01FFFFFFu) | (2u << 30);   // addr[56:32] | type=2
  u32x8 g1;
  g1[0] = (1u << 16)      // data_size = 2 bytes
        | (1u << 20)      // pad_enable
        | (4u << 22)      // pad_interval: 32 DWORDs (=128B of row data)
        | (3u << 25);     // pad_amount: 4 DWORDs (=16B -> 72-half LDS rows)
  g1[1] = 64u << 16;      // tensor_dim0[15:0]=64 at bits [63:48]
  g1[2] = 32u << 16;      // tensor_dim0 hi=0; tensor_dim1[15:0]=32 at [95:80]
  g1[3] = 64u << 16;      // tensor_dim1 hi=0; tile_dim0=64 at [127:112]
  g1[4] = 32u;            // tile_dim1=32; tile_dim2=0
  g1[5] = 1024u;          // tensor_dim0_stride = 1024 (elems), low 32 of 48
  g1[6] = 0u;             // stride hi; tensor_dim1_stride lo (unused, 2D)
  g1[7] = 0u;
  asm volatile("tensor_load_to_lds %0, %1" :: "s"(g0), "s"(g1) : "memory");
}

// Max-reduce across each 16-lane group using v_permlane16_b32 (VALU path; no
// LDS traffic). Selector nibbles: dest lane i reads lane i^d within its row.
__device__ __forceinline__ float max_red16(float v) {
  unsigned u, p;
  u = __float_as_uint(v);
  p = __builtin_amdgcn_permlane16(u, u, 0x67452301u, 0xEFCDAB89u, false, false); // xor 1
  v = fmaxf(v, __uint_as_float(p)); u = __float_as_uint(v);
  p = __builtin_amdgcn_permlane16(u, u, 0x54761032u, 0xDCFE98BAu, false, false); // xor 2
  v = fmaxf(v, __uint_as_float(p)); u = __float_as_uint(v);
  p = __builtin_amdgcn_permlane16(u, u, 0x32107654u, 0xBA98FEDCu, false, false); // xor 4
  v = fmaxf(v, __uint_as_float(p)); u = __float_as_uint(v);
  p = __builtin_amdgcn_permlane16(u, u, 0xFEDCBA98u, 0x76543210u, false, false); // xor 8
  return fmaxf(v, __uint_as_float(p));
}

union Frag16 {
  v16bf v;
  uint4 q[2];
  unsigned short h[16];
};

// A-matrix fragment, 16x32 bf16 (MxK), from row-major [m, k] with ushort stride ld.
// Layout (ISA 7.12.2): lanes 0-15 row m hold K {0..7, 16..23}; lanes 16-31 hold K {8..15, 24..31}.
__device__ __forceinline__ v16bf load_a_frag(const unsigned short* base, int ld, int lane) {
  const int m   = lane & 15;
  const int off = (lane < 16) ? 0 : 8;
  const unsigned short* p = base + m * ld + off;
  Frag16 f;
  f.q[0] = *(const uint4*)(p);
  f.q[1] = *(const uint4*)(p + 16);
  return f.v;
}

// B-matrix fragment, 32x16 bf16 (KxN), where B = Bn^T and Bn is row-major [n, k] (stride ld).
// Layout: lanes 0-15 col n hold K {0..7, 16..23}; lanes 16-31 hold K {8..15, 24..31}.
__device__ __forceinline__ v16bf load_bT_frag(const unsigned short* base, int ld, int lane) {
  const int n   = lane & 15;
  const int off = (lane < 16) ? 0 : 8;
  const unsigned short* p = base + n * ld + off;
  Frag16 f;
  f.q[0] = *(const uint4*)(p);
  f.q[1] = *(const uint4*)(p + 16);
  return f.v;
}

__device__ __forceinline__ v8f wmma_bf16(v16bf a, v16bf b, v8f c) {
  return __builtin_amdgcn_wmma_f32_16x16x32_bf16(false, a, false, b, (short)0, c, false, false);
}

// ---------------------------------------------------------------------------
// GEMM: C[M,N] = (A[M,K] @ Bn[N,K]^T + bias[N]) * outScale
// A is fp32 (A_F32, converted to bf16 during staging) or bf16 (async-copied);
// Bn is always fp32 weights; out fp32 or bf16.
// Block tile 128x128, 8 waves, wave tile 64x32 (4x2 WMMA 16x16 accumulators).
// ---------------------------------------------------------------------------
template <bool A_F32, bool OUT_F32>
__global__ __launch_bounds__(256) void gemm_kernel(
    const void* __restrict__ Aptr, const float* __restrict__ Bn,
    const float* __restrict__ bias, void* __restrict__ Cptr,
    int M, int N, int K, float outScale) {
  (void)M;
  __shared__ __align__(16) unsigned short As[128 * 40];   // 128 x 32, pad to 40
  __shared__ __align__(16) unsigned short Bs[128 * 40];

  const int tid    = threadIdx.x;
  const int lane   = tid & 31;
  const int w      = tid >> 5;      // 0..7
  const int waveM  = w >> 2;        // 0..1 -> 64-row half
  const int waveN  = w & 3;         // 0..3 -> 32-col quarter
  const int blockN = blockIdx.x * 128;
  const int blockM = blockIdx.y * 128;

  v8f acc[4][2];
#pragma unroll
  for (int mi = 0; mi < 4; ++mi)
#pragma unroll
    for (int ni = 0; ni < 2; ++ni)
#pragma unroll
      for (int j = 0; j < 8; ++j) acc[mi][ni][j] = 0.0f;

  for (int kk = 0; kk < K; kk += 32) {
    __syncthreads();
    // ---- stage A tile (128x32) -> bf16 LDS ----
    if constexpr (A_F32) {
      const float* Ag = (const float*)Aptr;
#pragma unroll
      for (int i = 0; i < 4; ++i) {
        int e = tid + i * 256;          // float4 chunk index, 1024 total
        int r = e >> 3;                 // 8 chunks per row
        int c = (e & 7) * 4;
        float4 f = *(const float4*)(Ag + (size_t)(blockM + r) * K + kk + c);
        ushort4 hh = make_ushort4(f2bf(f.x), f2bf(f.y), f2bf(f.z), f2bf(f.w));
        *(ushort4*)&As[r * 40 + c] = hh;
        if (kk + 32 < K)                // pull next A tile toward L2/WGP$
          __builtin_prefetch(Ag + (size_t)(blockM + r) * K + kk + 32 + c, 0, 1);
      }
    } else {
      const unsigned short* Ag = (const unsigned short*)Aptr;
#pragma unroll
      for (int i = 0; i < 2; ++i) {
        int e = tid + i * 256;          // 8-half chunk index, 512 total
        int r = e >> 2;                 // 4 chunks per row
        int c = (e & 3) * 8;
        // pure bf16 copy: use CDNA5 async global->LDS path (no VGPR round trip)
        async_copy_b128(&As[r * 40 + c],
                        Ag + (size_t)(blockM + r) * K + kk + c);
        if (kk + 32 < K)
          __builtin_prefetch(Ag + (size_t)(blockM + r) * K + kk + 32 + c, 0, 1);
      }
    }
    // ---- stage Bn tile (128x32 of [N,K] fp32 weights) -> bf16 LDS ----
#pragma unroll
    for (int i = 0; i < 4; ++i) {
      int e = tid + i * 256;
      int r = e >> 3;
      int c = (e & 7) * 4;
      float4 f = *(const float4*)(Bn + (size_t)(blockN + r) * K + kk + c);
      ushort4 hh = make_ushort4(f2bf(f.x), f2bf(f.y), f2bf(f.z), f2bf(f.w));
      *(ushort4*)&Bs[r * 40 + c] = hh;
      if (kk + 32 < K)
        __builtin_prefetch(Bn + (size_t)(blockN + r) * K + kk + 32 + c, 0, 1);
    }
    if constexpr (!A_F32) wait_async0();   // async LDS writes visible before barrier
    __syncthreads();

    v16bf afrag[4], bfrag[2];
#pragma unroll
    for (int mi = 0; mi < 4; ++mi)
      afrag[mi] = load_a_frag(&As[(waveM * 64 + mi * 16) * 40], 40, lane);
#pragma unroll
    for (int ni = 0; ni < 2; ++ni)
      bfrag[ni] = load_bT_frag(&Bs[(waveN * 32 + ni * 16) * 40], 40, lane);
#pragma unroll
    for (int mi = 0; mi < 4; ++mi)
#pragma unroll
      for (int ni = 0; ni < 2; ++ni)
        acc[mi][ni] = wmma_bf16(afrag[mi], bfrag[ni], acc[mi][ni]);
  }

  // ---- epilogue: (+bias)*scale, store (C layout: lane n = lane&15, m = j + 8*(lane>>4)) ----
  const int halfh = lane >> 4;
  const int nlane = lane & 15;
#pragma unroll
  for (int ni = 0; ni < 2; ++ni) {
    const int col = blockN + waveN * 32 + ni * 16 + nlane;
    const float bv = bias[col];
#pragma unroll
    for (int mi = 0; mi < 4; ++mi) {
#pragma unroll
      for (int j = 0; j < 8; ++j) {
        const int row = blockM + waveM * 64 + mi * 16 + j + 8 * halfh;
        const float val = (acc[mi][ni][j] + bv) * outScale;
        if constexpr (OUT_F32)
          ((float*)Cptr)[(size_t)row * N + col] = val;
        else
          ((unsigned short*)Cptr)[(size_t)row * N + col] = f2bf(val);
      }
    }
  }
}

// ---------------------------------------------------------------------------
// Flash attention: per block 128 q-rows of one (b,h); each wave owns 16 rows.
// kv processed in chunks of 32; online softmax; all matmuls via WMMA bf16.
// K chunk staged by the Tensor Data Mover (hardware LDS row padding); V chunk
// transposed through VGPRs. Q pre-scaled by 1/sqrt(Dh). Row sums of P via a
// WMMA against an all-ones B matrix; row max via v_permlane16 butterflies.
// Q/K/V/Ctx are bf16 [B*S, D], head slice = columns [h*64, h*64+64).
// ---------------------------------------------------------------------------
__global__ __launch_bounds__(256) void attn_kernel(
    const unsigned short* __restrict__ Q, const unsigned short* __restrict__ Kt,
    const unsigned short* __restrict__ V, unsigned short* __restrict__ Ctx) {
  __shared__ __align__(16) unsigned short Ks[32 * 72];        // K chunk [kv][dh], TDM-padded 64->72
  __shared__ __align__(16) unsigned short Vt[64 * 40];        // V chunk transposed [dh][kv], pad 32->40
  __shared__ __align__(16) unsigned short Pb[8 * 16 * 40];    // per-wave P tile [m][kv], pad 32->40

  const int tid  = threadIdx.x;
  const int lane = tid & 31;
  const int w    = tid >> 5;
  const int bh   = blockIdx.y;            // 0..63
  const int b    = bh >> 4;
  const int h    = bh & 15;
  const int qrow0 = blockIdx.x * 128 + w * 16;   // first q row (s index) of this wave

  const unsigned short* qbase = Q + ((size_t)(b * SEQ + qrow0)) * D_MODEL + h * HEAD_DIM;
  const v16bf qfrag0 = load_a_frag(qbase, D_MODEL, lane);        // dh 0..31
  const v16bf qfrag1 = load_a_frag(qbase + 32, D_MODEL, lane);   // dh 32..63

  Frag16 onesf;                            // all-ones bf16 B fragment for row sums
#pragma unroll
  for (int i = 0; i < 16; ++i) onesf.h[i] = 0x3F80u;

  v8f ctx[4];
#pragma unroll
  for (int t = 0; t < 4; ++t)
#pragma unroll
    for (int j = 0; j < 8; ++j) ctx[t][j] = 0.0f;

  float mrow[8], lrow[8];
#pragma unroll
  for (int j = 0; j < 8; ++j) { mrow[j] = -__builtin_inff(); lrow[j] = 0.0f; }

  unsigned short* P = &Pb[w * 16 * 40];

  for (int kv = 0; kv < SEQ; kv += 32) {
    __syncthreads();
    // ---- stage K chunk via TDM (one DMA op, wave 0) ----
    if (w == 0) {
      const unsigned short* ksrc =
          Kt + ((size_t)(b * SEQ + kv)) * D_MODEL + h * HEAD_DIM;
      tdm_load_k_chunk((unsigned)(uintptr_t)&Ks[0], ksrc);
      __builtin_amdgcn_s_wait_tensorcnt(0);
    }
    // ---- stage V chunk transposed via VGPRs (all threads) ----
    {
      const int r = tid >> 3;            // kv-local row 0..31
      const int c = (tid & 7) * 8;       // dh 0..56 step 8
      const size_t gro = ((size_t)(b * SEQ + kv + r)) * D_MODEL + h * HEAD_DIM + c;
      uint4 vv = *(const uint4*)(V + gro);
      const unsigned short* vh = (const unsigned short*)&vv;
#pragma unroll
      for (int i = 0; i < 8; ++i) Vt[(c + i) * 40 + r] = vh[i];
      if (kv + 32 < SEQ) {               // prefetch next chunk toward L2/WGP$
        const size_t gnx = gro + 32 * D_MODEL;
        __builtin_prefetch(Kt + gnx, 0, 1);
        __builtin_prefetch(V + gnx, 0, 1);
      }
    }
    __syncthreads();

    // ---- scores: two 16x16 tiles covering kv-local 0..15 and 16..31 ----
    v8f s0, s1;
#pragma unroll
    for (int j = 0; j < 8; ++j) { s0[j] = 0.0f; s1[j] = 0.0f; }
    {
      const v16bf k0a = load_bT_frag(&Ks[0 * 72 + 0],  72, lane);
      const v16bf k0b = load_bT_frag(&Ks[0 * 72 + 32], 72, lane);
      const v16bf k1a = load_bT_frag(&Ks[16 * 72 + 0],  72, lane);
      const v16bf k1b = load_bT_frag(&Ks[16 * 72 + 32], 72, lane);
      s0 = wmma_bf16(qfrag0, k0a, s0);
      s0 = wmma_bf16(qfrag1, k0b, s0);
      s1 = wmma_bf16(qfrag0, k1a, s1);
      s1 = wmma_bf16(qfrag1, k1b, s1);
    }

    // ---- online softmax max-update (rows per 16-lane half: m = j + 8*(lane>>4)) ----
    float fac[8];
#pragma unroll
    for (int j = 0; j < 8; ++j) {
      const float mx   = max_red16(fmaxf(s0[j], s1[j]));   // VALU butterfly
      const float mnew = fmaxf(mrow[j], mx);
      fac[j]  = __expf(mrow[j] - mnew);
      mrow[j] = mnew;
      s0[j] = __expf(s0[j] - mnew);      // P entries, fp32
      s1[j] = __expf(s1[j] - mnew);
    }
#pragma unroll
    for (int t = 0; t < 4; ++t)
#pragma unroll
      for (int j = 0; j < 8; ++j) ctx[t][j] *= fac[j];

    // ---- C-layout -> A-layout for P via per-wave LDS round trip ----
    {
      const int m0 = 8 * (lane >> 4);
      const int n  = lane & 15;
#pragma unroll
      for (int j = 0; j < 8; ++j) {
        P[(m0 + j) * 40 + n]      = f2bf(s0[j]);
        P[(m0 + j) * 40 + 16 + n] = f2bf(s1[j]);
      }
    }
    asm volatile("s_wait_dscnt 0" ::: "memory");   // same-wave LDS RAW

    const v16bf pfrag = load_a_frag(P, 40, lane);

    // row sums of (bf16-rounded) P via WMMA against ones: D[m][n] = rowsum(m),
    // replicated across lanes -> exactly the lrow update, no shuffles needed.
    v8f srow;
#pragma unroll
    for (int j = 0; j < 8; ++j) srow[j] = 0.0f;
    srow = wmma_bf16(pfrag, onesf.v, srow);
#pragma unroll
    for (int j = 0; j < 8; ++j) lrow[j] = lrow[j] * fac[j] + srow[j];

#pragma unroll
    for (int ns = 0; ns < 4; ++ns) {
      // B[k=kv][n=dh] = Vt[dh][kv]: lane n -> dh row of Vt, contiguous kv slice
      const v16bf vfrag = load_bT_frag(&Vt[(ns * 16) * 40], 40, lane);
      ctx[ns] = wmma_bf16(pfrag, vfrag, ctx[ns]);
    }
  }

  // ---- normalize and store context (bf16, [B*S, D] head slice) ----
  const int halfh = lane >> 4;
  const int nlane = lane & 15;
  float inv[8];
#pragma unroll
  for (int j = 0; j < 8; ++j) inv[j] = 1.0f / lrow[j];
#pragma unroll
  for (int ns = 0; ns < 4; ++ns) {
    const int dh = ns * 16 + nlane;
#pragma unroll
    for (int j = 0; j < 8; ++j) {
      const int m = j + 8 * halfh;
      const size_t o = ((size_t)(b * SEQ + qrow0 + m)) * D_MODEL + h * HEAD_DIM + dh;
      Ctx[o] = f2bf(ctx[ns][j] * inv[j]);
    }
  }
}

// ---------------------------------------------------------------------------
extern "C" void kernel_launch(void* const* d_in, const int* in_sizes, int n_in,
                              void* d_out, int out_size, void* d_ws, size_t ws_size,
                              hipStream_t stream) {
  (void)in_sizes; (void)n_in; (void)out_size; (void)ws_size;
  const float* x  = (const float*)d_in[0];
  const float* Wq = (const float*)d_in[1];
  const float* bq = (const float*)d_in[2];
  const float* Wk = (const float*)d_in[3];
  const float* bk = (const float*)d_in[4];
  const float* Wv = (const float*)d_in[5];
  const float* bv = (const float*)d_in[6];
  const float* Wo = (const float*)d_in[7];
  const float* bo = (const float*)d_in[8];
  float* out = (float*)d_out;

  unsigned short* ws = (unsigned short*)d_ws;
  const size_t BUF = (size_t)MROWS * D_MODEL;   // 8388608 bf16 elems = 16 MB
  unsigned short* Qb = ws;
  unsigned short* Kb = ws + BUF;
  unsigned short* Vb = ws + 2 * BUF;
  unsigned short* Cb = ws + 3 * BUF;

  const dim3 block(256);
  const dim3 gridG(D_MODEL / 128, MROWS / 128);     // (8, 64)
  const dim3 gridA(SEQ / 128, BATCH * N_HEADS);     // (16, 64)

  // Q projection pre-scaled by 1/sqrt(HEAD_DIM) = 0.125 (exact in bf16)
  gemm_kernel<true,  false><<<gridG, block, 0, stream>>>(x,  Wq, bq, Qb, MROWS, D_MODEL, D_MODEL, 0.125f);
  gemm_kernel<true,  false><<<gridG, block, 0, stream>>>(x,  Wk, bk, Kb, MROWS, D_MODEL, D_MODEL, 1.0f);
  gemm_kernel<true,  false><<<gridG, block, 0, stream>>>(x,  Wv, bv, Vb, MROWS, D_MODEL, D_MODEL, 1.0f);
  attn_kernel<<<gridA, block, 0, stream>>>(Qb, Kb, Vb, Cb);
  gemm_kernel<false, true ><<<gridG, block, 0, stream>>>(Cb, Wo, bo, out, MROWS, D_MODEL, D_MODEL, 1.0f);
}
